// MultiHeaded2MultiheadAttentionLSTM_1365799600443
// MI455X (gfx1250) — matrix operand order repaired and checked by reading, not compiled, not run on hardware
//
#include <hip/hip_runtime.h>

// ---------------------------------------------------------------------------
// Types for CDNA5 WMMA (gfx1250, wave32)
// ---------------------------------------------------------------------------
typedef __bf16 bf16;
typedef __attribute__((ext_vector_type(8)))  __bf16 v8bf;
typedef __attribute__((ext_vector_type(16))) __bf16 v16bf;
typedef __attribute__((ext_vector_type(8)))  float  v8f;

__device__ __forceinline__ float bf2f(bf16 x) {
    unsigned short s = __builtin_bit_cast(unsigned short, x);
    unsigned int u = ((unsigned int)s) << 16;
    return __builtin_bit_cast(float, u);
}
__device__ __forceinline__ bf16 f2bf(float f) {
    unsigned int u = __builtin_bit_cast(unsigned int, f);
    u += 0x7fffu + ((u >> 16) & 1u);           // round-to-nearest-even
    unsigned short s = (unsigned short)(u >> 16);
    return __builtin_bit_cast(bf16, s);
}
__device__ __forceinline__ float sigf(float x)  { return 1.f / (1.f + __expf(-x)); }
__device__ __forceinline__ float tanhf_(float x){ return 2.f / (1.f + __expf(-2.f * x)) - 1.f; }

// ---------------------------------------------------------------------------
// Elementwise f32 -> bf16 cast
// ---------------------------------------------------------------------------
__global__ void k_cast_bf16(const float* __restrict__ s, bf16* __restrict__ d, long n) {
    long i  = (long)blockIdx.x * blockDim.x + threadIdx.x;
    long st = (long)gridDim.x * blockDim.x;
    for (; i < n; i += st) d[i] = f2bf(s[i]);
}

// ---------------------------------------------------------------------------
// Generic batched WMMA GEMM:  C[M,N] = act(A[M,K] @ W[N,K]^T + b1 + b2)
// A, W bf16 row-major.
// 8 waves per block, each wave computes a 2x2 grid of 16x16 WMMA tiles
// (32 rows x 32 cols per wave), block tile = 128M x 64N. The 2x2 register
// blocking means 8 b128 loads feed 4 v_wmma ops (2x operand reuse vs 1x1).
// A-fragment (16-bit A 16x32): lane l<16 row=l   holds K {k0..k0+7, k0+16..k0+23}
//                              lane l>=16 row=l-16 holds +8 offset on both.
// B-fragment (16-bit B 32x16): lane l<16 col=l   holds K {k0..k0+7, k0+16..k0+23}
//                              lane l>=16 col=l-16 holds +8 offset on both.
// C layout: vgpr r: lane<16 -> (row r, col lane); lane>=16 -> (row r+8, col lane-16)
// ---------------------------------------------------------------------------
__global__ __launch_bounds__(256) void k_gemm(
    const bf16* __restrict__ A, long sA,
    const bf16* __restrict__ W, long sW,
    const float* __restrict__ b1, const float* __restrict__ b2,
    float* __restrict__ C, bf16* __restrict__ Cb, long sC,
    int M, int N, int K, int act)
{
    const int batch = blockIdx.z;
    A += (long)batch * sA;
    W += (long)batch * sW;
    const long cbase = (long)batch * sC;

    const int wave = threadIdx.x >> 5;
    const int lane = threadIdx.x & 31;
    const int l15  = lane & 15;
    const int hi   = lane >> 4;

    const int mt0 = (blockIdx.y * 8 + (wave & 3) * 2) * 16;   // 2 m-subtiles
    const int nt0 = (blockIdx.x * 4 + (wave >> 2) * 2) * 16;  // 2 n-subtiles

    int arow0 = mt0 + l15;      if (arow0 >= M) arow0 = M - 1;   // clamp
    int arow1 = mt0 + 16 + l15; if (arow1 >= M) arow1 = M - 1;
    const bf16* Ap0 = A + (long)arow0 * K + hi * 8;
    const bf16* Ap1 = A + (long)arow1 * K + hi * 8;
    const bf16* Wp0 = W + (long)(nt0 + l15) * K + hi * 8;
    const bf16* Wp1 = W + (long)(nt0 + 16 + l15) * K + hi * 8;

    v8f acc00, acc01, acc10, acc11;
    #pragma unroll
    for (int i = 0; i < 8; ++i) { acc00[i] = 0.f; acc01[i] = 0.f; acc10[i] = 0.f; acc11[i] = 0.f; }

    for (int k0 = 0; k0 < K; k0 += 32) {
        v8bf a0lo = *(const v8bf*)(Ap0 + k0);
        v8bf a0hi = *(const v8bf*)(Ap0 + k0 + 16);
        v8bf a1lo = *(const v8bf*)(Ap1 + k0);
        v8bf a1hi = *(const v8bf*)(Ap1 + k0 + 16);
        v8bf b0lo = *(const v8bf*)(Wp0 + k0);
        v8bf b0hi = *(const v8bf*)(Wp0 + k0 + 16);
        v8bf b1lo = *(const v8bf*)(Wp1 + k0);
        v8bf b1hi = *(const v8bf*)(Wp1 + k0 + 16);
        v16bf a0, a1, b0, b1;
        #pragma unroll
        for (int i = 0; i < 8; ++i) {
            a0[i] = a0lo[i]; a0[i + 8] = a0hi[i];
            a1[i] = a1lo[i]; a1[i + 8] = a1hi[i];
            b0[i] = b0lo[i]; b0[i + 8] = b0hi[i];
            b1[i] = b1lo[i]; b1[i + 8] = b1hi[i];
        }
        acc00 = __builtin_amdgcn_wmma_f32_16x16x32_bf16(false, a0, false, b0, (short)0, acc00, false, false);
        acc01 = __builtin_amdgcn_wmma_f32_16x16x32_bf16(false, a0, false, b1, (short)0, acc01, false, false);
        acc10 = __builtin_amdgcn_wmma_f32_16x16x32_bf16(false, a1, false, b0, (short)0, acc10, false, false);
        acc11 = __builtin_amdgcn_wmma_f32_16x16x32_bf16(false, a1, false, b1, (short)0, acc11, false, false);
    }

    const int col0 = nt0 + l15;
    const int col1 = nt0 + 16 + l15;
    float bia0 = 0.f, bia1 = 0.f;
    if (b1) { bia0 += b1[col0]; bia1 += b1[col1]; }
    if (b2) { bia0 += b2[col0]; bia1 += b2[col1]; }

    #pragma unroll
    for (int r = 0; r < 8; ++r) {
        int row0 = mt0 + r + hi * 8;
        int row1 = mt0 + 16 + r + hi * 8;
        if (row0 < M) {
            float v = acc00[r] + bia0;
            if (act == 1) v = v > 0.f ? v : 0.f;
            long idx = cbase + (long)row0 * N + col0;
            if (C)  C[idx]  = v;
            if (Cb) Cb[idx] = f2bf(v);
            float w = acc01[r] + bia1;
            if (act == 1) w = w > 0.f ? w : 0.f;
            idx = cbase + (long)row0 * N + col1;
            if (C)  C[idx]  = w;
            if (Cb) Cb[idx] = f2bf(w);
        }
        if (row1 < M) {
            float v = acc10[r] + bia0;
            if (act == 1) v = v > 0.f ? v : 0.f;
            long idx = cbase + (long)row1 * N + col0;
            if (C)  C[idx]  = v;
            if (Cb) Cb[idx] = f2bf(v);
            float w = acc11[r] + bia1;
            if (act == 1) w = w > 0.f ? w : 0.f;
            idx = cbase + (long)row1 * N + col1;
            if (C)  C[idx]  = w;
            if (Cb) Cb[idx] = f2bf(w);
        }
    }
}

// ---------------------------------------------------------------------------
// Persistent LSTM recurrence. Grid = 2 blocks (16 batch rows each), 1024 thr
// = 32 waves. Wave w owns hidden columns [16w, 16w+16) for ALL four gates, so
// the gate nonlinearity stays in registers. h lives in LDS as bf16; c lives
// in registers. preact = x@Wih^T + bih + bhh precomputed for all timesteps.
// Next timestep's preact rows are prefetched (global_prefetch_b8) behind the
// WMMA K-loop to hide L2 latency in the elementwise phase.
// ---------------------------------------------------------------------------
__global__ __launch_bounds__(1024) void k_lstm(
    const float* __restrict__ pre,   // [B,S,2048] i,f,g,o
    const bf16*  __restrict__ Whh,   // [2048,512]
    const float* __restrict__ h0, const float* __restrict__ c0,   // [B,512] or null
    float* __restrict__ hs, bf16* __restrict__ hsb,               // [B,S,512] or null
    float* __restrict__ hN, float* __restrict__ cN,               // [B,512] or null
    int S)
{
    __shared__ bf16 hsm[16][512];                 // 16 KB

    const int tid  = threadIdx.x;
    const int wave = tid >> 5;
    const int lane = tid & 31;
    const int l15  = lane & 15;
    const int hi   = lane >> 4;
    const int bb   = blockIdx.x * 16;             // batch base
    const int j0   = wave * 16;                   // hidden-column tile

    // init h in LDS (bf16), c in registers (f32)
    for (int i = tid; i < 16 * 512; i += 1024) {
        int r = i >> 9, c = i & 511;
        float hv = h0 ? h0[(long)(bb + r) * 512 + c] : 0.f;
        hsm[r][c] = f2bf(hv);
    }
    v8f creg;
    #pragma unroll
    for (int r = 0; r < 8; ++r) {
        int lr = r + hi * 8;
        creg[r] = c0 ? c0[(long)(bb + lr) * 512 + j0 + l15] : 0.f;
    }
    __syncthreads();

    for (int t = 0; t < S; ++t) {
        // prefetch next step's preact (one address per gate chunk, r=0 rows)
        if (t + 1 < S) {
            long pb = ((long)(bb + hi * 8) * S + (t + 1)) * 2048 + j0 + l15;
            __builtin_prefetch(pre + pb, 0, 0);
            __builtin_prefetch(pre + pb + 512, 0, 0);
            __builtin_prefetch(pre + pb + 1024, 0, 0);
            __builtin_prefetch(pre + pb + 1536, 0, 0);
        }

        v8f acc[4];
        #pragma unroll
        for (int g = 0; g < 4; ++g)
            #pragma unroll
            for (int i = 0; i < 8; ++i) acc[g][i] = 0.f;

        for (int k0 = 0; k0 < 512; k0 += 32) {
            v8bf alo = *(const v8bf*)&hsm[l15][hi * 8 + k0];
            v8bf ahi = *(const v8bf*)&hsm[l15][hi * 8 + k0 + 16];
            v16bf a;
            #pragma unroll
            for (int i = 0; i < 8; ++i) { a[i] = alo[i]; a[i + 8] = ahi[i]; }
            #pragma unroll
            for (int g = 0; g < 4; ++g) {
                const bf16* wp = Whh + (long)(g * 512 + j0 + l15) * 512 + hi * 8 + k0;
                v8bf blo = *(const v8bf*)wp;
                v8bf bhi = *(const v8bf*)(wp + 16);
                v16bf b;
                #pragma unroll
                for (int i = 0; i < 8; ++i) { b[i] = blo[i]; b[i + 8] = bhi[i]; }
                acc[g] = __builtin_amdgcn_wmma_f32_16x16x32_bf16(false, a, false, b,
                                                                 (short)0, acc[g], false, false);
            }
        }

        float hh[8], cv[8];
        #pragma unroll
        for (int r = 0; r < 8; ++r) {
            int lr = r + hi * 8;
            long bI = bb + lr;
            long base = ((long)bI * S + t) * 2048 + j0 + l15;
            float gi = acc[0][r] + pre[base];
            float gf = acc[1][r] + pre[base + 512];
            float gg = acc[2][r] + pre[base + 1024];
            float go = acc[3][r] + pre[base + 1536];
            float cc = sigf(gf) * creg[r] + sigf(gi) * tanhf_(gg);
            creg[r] = cc;
            cv[r]   = cc;
            hh[r]   = sigf(go) * tanhf_(cc);
        }
        __syncthreads();  // all reads of hsm done
        #pragma unroll
        for (int r = 0; r < 8; ++r) {
            int lr  = r + hi * 8;
            int col = j0 + l15;
            hsm[lr][col] = f2bf(hh[r]);
            long o = ((long)(bb + lr) * S + t) * 512 + col;
            if (hs)  hs[o]  = hh[r];
            if (hsb) hsb[o] = f2bf(hh[r]);
            if (t == S - 1) {
                long q = (long)(bb + lr) * 512 + col;
                if (hN) hN[q] = hh[r];
                if (cN) cN[q] = cv[r];
            }
        }
        __syncthreads();  // new h visible
    }
}

// ---------------------------------------------------------------------------
// QKV split: [B*S, 1536] f32 -> Q,K [b,h,s,d] bf16 and V^T [b,h,d,s] bf16
// ---------------------------------------------------------------------------
__global__ void k_qkvsplit(const float* __restrict__ qkv,
                           bf16* __restrict__ Q, bf16* __restrict__ Kb, bf16* __restrict__ Vt) {
    const long n = 32L * 256 * 1536;
    long i  = (long)blockIdx.x * blockDim.x + threadIdx.x;
    long st = (long)gridDim.x * blockDim.x;
    for (; i < n; i += st) {
        long r = i / 1536; int c = (int)(i - r * 1536);
        int b = (int)(r >> 8), s = (int)(r & 255);
        int w = c / 512, wi = c - w * 512;
        int h = wi >> 6, d = wi & 63;
        bf16 v = f2bf(qkv[i]);
        long bh = (long)b * 8 + h;
        if (w == 0)      Q [(bh * 256 + s) * 64 + d] = v;
        else if (w == 1) Kb[(bh * 256 + s) * 64 + d] = v;
        else             Vt[(bh * 64 + d) * 256 + s] = v;
    }
}

// ctx [b,h,s,d] f32 -> [b,s, h*64+d] bf16
__global__ void k_ctxmerge(const float* __restrict__ in, bf16* __restrict__ out) {
    const long n = 32L * 8 * 256 * 64;
    long i  = (long)blockIdx.x * blockDim.x + threadIdx.x;
    long st = (long)gridDim.x * blockDim.x;
    for (; i < n; i += st) {
        int b = (int)(i >> 17);
        int rem = (int)(i & 131071);
        int h = rem >> 14;
        int rem2 = rem & 16383;
        int s = rem2 >> 6, d = rem2 & 63;
        out[((long)b * 256 + s) * 512 + h * 64 + d] = f2bf(in[i]);
    }
}

// row softmax over 256 cols with 1/sqrt(64) scale; writes bf16 probs
__global__ __launch_bounds__(256) void k_softmax(const float* __restrict__ s, bf16* __restrict__ p) {
    __shared__ float red[256];
    const long row = blockIdx.x;
    float v = s[row * 256 + threadIdx.x] * 0.125f;
    red[threadIdx.x] = v; __syncthreads();
    for (int o = 128; o > 0; o >>= 1) {
        if (threadIdx.x < o) red[threadIdx.x] = fmaxf(red[threadIdx.x], red[threadIdx.x + o]);
        __syncthreads();
    }
    float m = red[0]; __syncthreads();
    float e = __expf(v - m);
    red[threadIdx.x] = e; __syncthreads();
    for (int o = 128; o > 0; o >>= 1) {
        if (threadIdx.x < o) red[threadIdx.x] += red[threadIdx.x + o];
        __syncthreads();
    }
    p[row * 256 + threadIdx.x] = f2bf(e / red[0]);
}

// y = LN(a + res) * g + b over rows of 512; optional f32 / bf16 outputs
__global__ __launch_bounds__(256) void k_addln(
    const float* __restrict__ a, const float* __restrict__ res,
    const float* __restrict__ g, const float* __restrict__ bt,
    float* __restrict__ of, bf16* __restrict__ ob)
{
    __shared__ float s1[256], s2[256];
    const long row = blockIdx.x;
    const float* ar = a + row * 512;
    const float* rr = res + row * 512;
    float x0 = ar[threadIdx.x]       + rr[threadIdx.x];
    float x1 = ar[threadIdx.x + 256] + rr[threadIdx.x + 256];
    s1[threadIdx.x] = x0 + x1;
    s2[threadIdx.x] = x0 * x0 + x1 * x1;
    __syncthreads();
    for (int o = 128; o > 0; o >>= 1) {
        if (threadIdx.x < o) { s1[threadIdx.x] += s1[threadIdx.x + o]; s2[threadIdx.x] += s2[threadIdx.x + o]; }
        __syncthreads();
    }
    float mean = s1[0] * (1.f / 512.f);
    float var  = s2[0] * (1.f / 512.f) - mean * mean;
    float rs   = rsqrtf(var + 1e-5f);
    {
        int c = threadIdx.x;
        float y = (x0 - mean) * rs * g[c] + bt[c];
        if (of) of[row * 512 + c] = y;
        if (ob) ob[row * 512 + c] = f2bf(y);
        c = threadIdx.x + 256;
        float y1 = (x1 - mean) * rs * g[c] + bt[c];
        if (of) of[row * 512 + c] = y1;
        if (ob) ob[row * 512 + c] = f2bf(y1);
    }
}

// concat final hidden states -> bf16 [32, 1024]
__global__ void k_concat(const float* __restrict__ h1, const float* __restrict__ h2,
                         bf16* __restrict__ cat) {
    int i = blockIdx.x * blockDim.x + threadIdx.x;
    if (i >= 32 * 1024) return;
    int b = i >> 10, c = i & 1023;
    float v = (c < 512) ? h1[b * 512 + c] : h2[b * 512 + (c - 512)];
    cat[i] = f2bf(v);
}

// ---------------------------------------------------------------------------
// Host driver
// ---------------------------------------------------------------------------
extern "C" void kernel_launch(void* const* d_in, const int* in_sizes, int n_in,
                              void* d_out, int out_size, void* d_ws, size_t ws_size,
                              hipStream_t stream) {
    (void)in_sizes; (void)n_in; (void)out_size; (void)ws_size;

    const float* fc   = (const float*)d_in[0];
    const float* mg   = (const float*)d_in[1];
    const float* Wih  = (const float*)d_in[2];   // [6,2048,512]
    const float* Whh  = (const float*)d_in[3];   // [6,2048,512]
    const float* bih  = (const float*)d_in[4];   // [6,2048]
    const float* bhh  = (const float*)d_in[5];   // [6,2048]
    const float* inw  = (const float*)d_in[6];   // [2,1536,512]
    const float* inb  = (const float*)d_in[7];   // [2,1536]
    const float* outw = (const float*)d_in[8];   // [2,512,512]
    const float* outb = (const float*)d_in[9];   // [2,512]
    const float* lng  = (const float*)d_in[10];  // [2,512]
    const float* lnb  = (const float*)d_in[11];  // [2,512]
    const float* mw1  = (const float*)d_in[12];  // [2,2048,512]
    const float* mb1  = (const float*)d_in[13];  // [2,2048]
    const float* mw2  = (const float*)d_in[14];  // [2,512,2048]
    const float* mb2  = (const float*)d_in[15];  // [2,512]
    const float* fw1  = (const float*)d_in[16];  // [2048,1024]
    const float* fb1  = (const float*)d_in[17];  // [2048]
    const float* fw2  = (const float*)d_in[18];  // [512,2048]
    const float* fb2  = (const float*)d_in[19];  // [512]

    char* ws = (char*)d_ws;
    size_t off = 0;
    auto alloc = [&](size_t bytes) -> void* {
        off = (off + 255) & ~(size_t)255;
        void* p = ws + off;
        off += bytes;
        return p;
    };

    // bf16 weight copies (L2-resident, reused every GEMM / timestep)
    bf16* wihB  = (bf16*)alloc(6L * 2048 * 512 * 2);
    bf16* whhB  = (bf16*)alloc(6L * 2048 * 512 * 2);
    bf16* inwB  = (bf16*)alloc(2L * 1536 * 512 * 2);
    bf16* outwB = (bf16*)alloc(2L * 512 * 512 * 2);
    bf16* mw1B  = (bf16*)alloc(2L * 2048 * 512 * 2);
    bf16* mw2B  = (bf16*)alloc(2L * 512 * 2048 * 2);
    bf16* fw1B  = (bf16*)alloc(2048L * 1024 * 2);
    bf16* fw2B  = (bf16*)alloc(512L * 2048 * 2);

    // activation buffers (reused across the two branches)
    bf16*  xB    = (bf16*) alloc(8192L * 512 * 2);
    float* gates = (float*)alloc(8192L * 2048 * 4);
    float* tmpf  = (float*)alloc(8192L * 512 * 4);     // out-proj pre-LN tmp
    bf16*  hsB   = (bf16*) alloc(8192L * 512 * 2);     // lstm hidden seq (bf16)
    float* qkv   = (float*)alloc(8192L * 1536 * 4);    // reused as out2 preact region
    bf16*  qB    = (bf16*) alloc(32L * 8 * 256 * 64 * 2);
    bf16*  kB    = (bf16*) alloc(32L * 8 * 256 * 64 * 2);
    bf16*  vtB   = (bf16*) alloc(32L * 8 * 64 * 256 * 2);
    float* scor  = (float*)alloc(32L * 8 * 256 * 256 * 4);
    bf16*  probs = (bf16*) alloc(32L * 8 * 256 * 256 * 2);  // aliased as mlp1 bf16
    float* ctxF  = (float*)alloc(32L * 8 * 256 * 64 * 4);
    bf16*  ctxB  = (bf16*) alloc(8192L * 512 * 2);
    float* attnF = (float*)alloc(8192L * 512 * 4);
    bf16*  attnB = (bf16*) alloc(8192L * 512 * 2);
    bf16*  out2B = (bf16*) alloc(8192L * 512 * 2);
    float* hA    = (float*)alloc(32L * 512 * 4);
    float* cA    = (float*)alloc(32L * 512 * 4);
    float* h1f   = (float*)alloc(32L * 512 * 4);
    float* h2f   = (float*)alloc(32L * 512 * 4);
    bf16*  catB  = (bf16*) alloc(32L * 1024 * 2);
    bf16*  fc1B  = (bf16*) alloc(32L * 2048 * 2);

    bf16*  mlp1B   = probs;          // 8192*2048 bf16 == 32*8*256*256 bf16
    float* out2pre = qkv;            // 16.8 MB within 50 MB qkv region

    auto cast = [&](const float* s, bf16* d, long n) {
        k_cast_bf16<<<2048, 256, 0, stream>>>(s, d, n);
    };
    cast(Wih,  wihB,  6L * 2048 * 512);
    cast(Whh,  whhB,  6L * 2048 * 512);
    cast(inw,  inwB,  2L * 1536 * 512);
    cast(outw, outwB, 2L * 512 * 512);
    cast(mw1,  mw1B,  2L * 2048 * 512);
    cast(mw2,  mw2B,  2L * 512 * 2048);
    cast(fw1,  fw1B,  2048L * 1024);
    cast(fw2,  fw2B,  512L * 2048);

    auto gemm = [&](const bf16* A, long sA, const bf16* W, long sW,
                    const float* b1, const float* b2,
                    float* C, bf16* Cb, long sC,
                    int M, int N, int K, int act, int nb) {
        dim3 g((unsigned)(N / 64), (unsigned)((M + 127) / 128), (unsigned)nb);
        k_gemm<<<g, 256, 0, stream>>>(A, sA, W, sW, b1, b2, C, Cb, sC, M, N, K, act);
    };

    auto branch = [&](const float* x, int ei, int c1, int c2, int ai, int li, float* hout) {
        // enc LSTM
        cast(x, xB, 8192L * 512);
        gemm(xB, 0, wihB + (long)ei * 2048 * 512, 0, bih + ei * 2048, bhh + ei * 2048,
             gates, nullptr, 0, 8192, 2048, 512, 0, 1);
        k_lstm<<<2, 1024, 0, stream>>>(gates, whhB + (long)ei * 2048 * 512,
                                       nullptr, nullptr, nullptr, hsB, hA, cA, 256);
        // MHA
        gemm(hsB, 0, inwB + (long)ai * 1536 * 512, 0, inb + ai * 1536, nullptr,
             qkv, nullptr, 0, 8192, 1536, 512, 0, 1);
        k_qkvsplit<<<4096, 256, 0, stream>>>(qkv, qB, kB, vtB);
        gemm(qB, 16384, kB, 16384, nullptr, nullptr, scor, nullptr, 65536,
             256, 256, 64, 0, 256);
        k_softmax<<<65536, 256, 0, stream>>>(scor, probs);
        gemm(probs, 65536, vtB, 16384, nullptr, nullptr, ctxF, nullptr, 16384,
             256, 64, 256, 0, 256);
        k_ctxmerge<<<4096, 256, 0, stream>>>(ctxF, ctxB);
        gemm(ctxB, 0, outwB + (long)ai * 512 * 512, 0, outb + ai * 512, nullptr,
             tmpf, nullptr, 0, 8192, 512, 512, 0, 1);
        k_addln<<<8192, 256, 0, stream>>>(tmpf, x, lng + li * 512, lnb + li * 512,
                                          attnF, attnB);
        // LSTM 2 (carries state)
        gemm(attnB, 0, wihB + (long)c1 * 2048 * 512, 0, bih + c1 * 2048, bhh + c1 * 2048,
             gates, nullptr, 0, 8192, 2048, 512, 0, 1);
        k_lstm<<<2, 1024, 0, stream>>>(gates, whhB + (long)c1 * 2048 * 512,
                                       hA, cA, nullptr, hsB, hA, cA, 256);
        // MLP + LN
        gemm(hsB, 0, mw1B + (long)li * 2048 * 512, 0, mb1 + li * 2048, nullptr,
             nullptr, mlp1B, 0, 8192, 2048, 512, 1, 1);
        gemm(mlp1B, 0, mw2B + (long)li * 512 * 2048, 0, mb2 + li * 512, nullptr,
             out2pre, nullptr, 0, 8192, 512, 2048, 0, 1);
        k_addln<<<8192, 256, 0, stream>>>(out2pre, attnF, lng + li * 512, lnb + li * 512,
                                          nullptr, out2B);
        // LSTM 3 (final state only)
        gemm(out2B, 0, wihB + (long)c2 * 2048 * 512, 0, bih + c2 * 2048, bhh + c2 * 2048,
             gates, nullptr, 0, 8192, 2048, 512, 0, 1);
        k_lstm<<<2, 1024, 0, stream>>>(gates, whhB + (long)c2 * 2048 * 512,
                                       hA, cA, nullptr, nullptr, hout, nullptr, 256);
    };

    branch(fc, 0, 1, 2, 0, 0, h1f);
    branch(mg, 3, 4, 5, 1, 1, h2f);

    // head: concat -> relu FC -> FC -> d_out
    k_concat<<<128, 256, 0, stream>>>(h1f, h2f, catB);
    gemm(catB, 0, fw1B, 0, fb1, nullptr, nullptr, fc1B, 0, 32, 2048, 1024, 1, 1);
    gemm(fc1B, 0, fw2B, 0, fb2, nullptr, (float*)d_out, nullptr, 0, 32, 512, 2048, 0, 1);
}
